// BasicBlock_47837345742925
// MI455X (gfx1250) — compile-verified, matrix operand order checked by reading, not run on hardware
//
#include <hip/hip_runtime.h>

// ---------------------------------------------------------------------------
// norm-dist (L8) residual block for MI455X / gfx1250, wave32.
//   layer: y  = (sum_k |px - wc|^8)^(1/8)
//          dl = (sum_k max(pl-wc, wc-pu, 0)^8)^(1/8)
//          du = (sum_k max(|pl-wc|, |pu-wc|)^8)^(1/8)
//   wc = w - rowmean(w)  (the only matmul-shaped op -> WMMA row-sum vs ones)
// Data movement: x/l/u halo slabs staged into LDS with
// GLOBAL_LOAD_ASYNC_TO_LDS_B32 (ASYNCcnt), zero-padding via plain LDS stores.
// ---------------------------------------------------------------------------

typedef __attribute__((ext_vector_type(16))) _Float16 v16h;
typedef __attribute__((ext_vector_type(8)))  float    v8f;

#define NB   4
#define NC   32
#define NH   32
#define NW   32
#define KK   288                    // 32*3*3
#define NELEM (NB * NC * NH * NW)   // 131072

// Issue one async global->LDS B32 copy for the executing lane.
// vdst = per-lane LDS byte address (low 32 bits of the generic pointer),
// vaddr = 64-bit global address.  Tracked with ASYNCcnt.
__device__ __forceinline__ void async_copy_b32(const float* gptr, float* lptr) {
  unsigned           laddr = (unsigned)(unsigned long long)(uintptr_t)lptr;
  unsigned long long gaddr = (unsigned long long)(uintptr_t)gptr;
  asm volatile("global_load_async_to_lds_b32 %0, %1, off"
               :: "v"(laddr), "v"(gaddr)
               : "memory");
}

__device__ __forceinline__ void wait_async_all() {
  asm volatile("s_wait_asynccnt 0x0" ::: "memory");
}

// ---------------------------------------------------------------------------
// Weight centering: rowmean over K=288 computed on the matrix unit.
// One wave. For each 16-row tile: 9 x v_wmma_f32_16x16x32_f16 with B = ones
// gives D[m, n] = sum_k A[m, k] (broadcast over n). Exact element->K slot
// mapping inside a lane is irrelevant for a ones-B reduction: lanes m and m+16
// together just need to cover the 32 distinct K of row m per chunk.
// ---------------------------------------------------------------------------
__global__ void __launch_bounds__(32)
center_weights_wmma(const float* __restrict__ W1, const float* __restrict__ W2,
                    float* __restrict__ WC1, float* __restrict__ WC2) {
  __shared__ float smean[NC];
  const int lane = threadIdx.x;   // 0..31
  const int m    = lane & 15;
  const int hi   = lane >> 4;     // 0: lanes 0-15, 1: lanes 16-31

  v16h ones;
#pragma unroll
  for (int i = 0; i < 16; ++i) ones[i] = (_Float16)1.0f;

  for (int which = 0; which < 2; ++which) {
    const float* W  = which ? W2 : W1;
    float*       WC = which ? WC2 : WC1;

    for (int tile = 0; tile < 2; ++tile) {
      const float* wrow = W + (tile * 16 + m) * KK;
      v8f acc = {};
      for (int kc = 0; kc < 9; ++kc) {         // 9 * 32 = 288
        const int k0 = kc * 32 + hi * 8;
        v16h a;
#pragma unroll
        for (int i = 0; i < 8; ++i) {
          a[i]     = (_Float16)wrow[k0 + i];
          a[8 + i] = (_Float16)wrow[k0 + 16 + i];
        }
        // D = A(16x32) * ones(32x16) + C  ->  per-row sums broadcast over n
        acc = __builtin_amdgcn_wmma_f32_16x16x32_f16(
            /*neg_a=*/false, a, /*neg_b=*/false, ones,
            /*c_mod=*/(short)0, acc, /*reuse_a=*/false, /*reuse_b=*/false);
      }
      // VGPR r of lanes 0-15 holds row r, of lanes 16-31 holds row 8+r.
      if (lane == 0 || lane == 16) {
#pragma unroll
        for (int r = 0; r < 8; ++r) smean[tile * 16 + hi * 8 + r] = acc[r];
      }
    }
    __syncthreads();
    for (int idx = lane; idx < NC * KK; idx += 32) {
      WC[idx] = W[idx] - smean[idx / KK] * (1.0f / (float)KK);
    }
    __syncthreads();
  }
}

// ---------------------------------------------------------------------------
// One norm-dist layer (fused y/dl/du + optional residual + relu).
// Block: 256 threads = 8 waves; one output row (32 cols) per block.
// Wave w handles cout group w (4 couts), all 32 lanes on distinct cols:
//   - weight loads are wave-uniform (one cacheline, WGP$-resident, 36 KB total)
//   - LDS patch reads are stride-1 across lanes (conflict-free, 64 banks)
// LDS stages the 3-row halo slab of x/l/u for all 32 cin via async-to-LDS:
// 3*32*3*34*4 = 38 KB.
// ---------------------------------------------------------------------------
__global__ void __launch_bounds__(256)
norm_dist_layer(const float* __restrict__ X, const float* __restrict__ L,
                const float* __restrict__ U, const float* __restrict__ WC,
                const float* __restrict__ RX, const float* __restrict__ RL,
                const float* __restrict__ RU,
                float* __restrict__ OY, float* __restrict__ OL,
                float* __restrict__ OU) {
  __shared__ float sX[NC][3][NW + 2];
  __shared__ float sL[NC][3][NW + 2];
  __shared__ float sU[NC][3][NW + 2];

  const int tid = threadIdx.x;
  const int b   = blockIdx.x >> 5;       // 4 images
  const int h0  = blockIdx.x & 31;       // output row

  // ---- stage halo slab: async global->LDS, zero-pad borders with DS stores
  for (int idx = tid; idx < NC * 3 * (NW + 2); idx += 256) {
    const int cc  = idx % (NW + 2);
    const int t   = idx / (NW + 2);
    const int rr  = t % 3;
    const int cin = t / 3;
    const int gh  = h0 + rr - 1;
    const int gw  = cc - 1;
    const bool ok = (gh >= 0) & (gh < NH) & (gw >= 0) & (gw < NW);
    const int g   = ((b * NC + cin) * NH + gh) * NW + gw;
    if (ok) {
      async_copy_b32(X + g, &sX[cin][rr][cc]);
      async_copy_b32(L + g, &sL[cin][rr][cc]);
      async_copy_b32(U + g, &sU[cin][rr][cc]);
    } else {
      sX[cin][rr][cc] = 0.0f;
      sL[cin][rr][cc] = 0.0f;
      sU[cin][rr][cc] = 0.0f;
    }
  }
  wait_async_all();     // ASYNCcnt -> 0 for this wave's issued copies
  __syncthreads();      // then make the slab visible workgroup-wide

  const int col = tid & 31;   // lane -> column  (stride-1 LDS)
  const int cg  = tid >> 5;   // wave -> cout group (4 couts, wave-uniform)

  float accY[4], accL[4], accU[4];
#pragma unroll
  for (int i = 0; i < 4; ++i) { accY[i] = 0.f; accL[i] = 0.f; accU[i] = 0.f; }

  for (int cin = 0; cin < NC; ++cin) {
    float pxv[9], plv[9], puv[9];
    const float* bx = &sX[cin][0][col];
    const float* bl = &sL[cin][0][col];
    const float* bu = &sU[cin][0][col];
#pragma unroll
    for (int dh = 0; dh < 3; ++dh)
#pragma unroll
      for (int dw = 0; dw < 3; ++dw) {
        const int o = dh * (NW + 2) + dw;
        const int j = dh * 3 + dw;
        pxv[j] = bx[o]; plv[j] = bl[o]; puv[j] = bu[o];
      }
#pragma unroll
    for (int co = 0; co < 4; ++co) {
      const float* wr = WC + (cg * 4 + co) * KK + cin * 9;
#pragma unroll
      for (int j = 0; j < 9; ++j) {
        const float wv = wr[j];
        // y: |d|^8  (even power -> three squarings)
        const float d  = pxv[j] - wv;
        const float d2 = d * d, d4 = d2 * d2;
        accY[co] += d4 * d4;
        // dl: max(l-w, w-u, 0)^8
        const float t1  = plv[j] - wv;
        const float t2  = wv - puv[j];
        const float tl  = fmaxf(fmaxf(t1, t2), 0.0f);
        const float tl2 = tl * tl, tl4 = tl2 * tl2;
        accL[co] += tl4 * tl4;
        // du: max(|l-w|, |u-w|)^8   (|.| is a free src modifier)
        const float tu  = fmaxf(fabsf(t1), fabsf(t2));
        const float tu2 = tu * tu, tu4 = tu2 * tu2;
        accU[co] += tu4 * tu4;
      }
    }
  }

#pragma unroll
  for (int co = 0; co < 4; ++co) {
    const int cout = cg * 4 + co;
    const int g    = ((b * NC + cout) * NH + h0) * NW + col;
    // s^(1/8) = sqrt(sqrt(sqrt(s)))  -- three v_sqrt_f32 trans ops
    float yv = sqrtf(sqrtf(sqrtf(accY[co])));
    float lv = sqrtf(sqrtf(sqrtf(accL[co])));
    float uv = sqrtf(sqrtf(sqrtf(accU[co])));
    if (RX) { yv += RX[g]; lv += RL[g]; uv += RU[g]; }
    OY[g] = fmaxf(yv, 0.0f);
    OL[g] = fmaxf(lv, 0.0f);
    OU[g] = fmaxf(uv, 0.0f);
  }
}

// ---------------------------------------------------------------------------
extern "C" void kernel_launch(void* const* d_in, const int* in_sizes, int n_in,
                              void* d_out, int out_size, void* d_ws,
                              size_t ws_size, hipStream_t stream) {
  const float* x  = (const float*)d_in[0];
  const float* lo = (const float*)d_in[1];
  const float* up = (const float*)d_in[2];
  const float* w1 = (const float*)d_in[3];
  const float* w2 = (const float*)d_in[4];
  float* out = (float*)d_out;

  // workspace: centered weights (2 x 36KB) + layer-1 triple (3 x 512KB)
  float* wc1 = (float*)d_ws;
  float* wc2 = wc1 + NC * KK;
  float* y1  = wc2 + NC * KK;
  float* l1  = y1 + NELEM;
  float* u1  = l1 + NELEM;

  center_weights_wmma<<<1, 32, 0, stream>>>(w1, w2, wc1, wc2);

  // layer 1: no residual, relu
  norm_dist_layer<<<NB * NH, 256, 0, stream>>>(
      x, lo, up, wc1, nullptr, nullptr, nullptr, y1, l1, u1);

  // layer 2: residual = original (x, lower, upper), relu; write stacked output
  norm_dist_layer<<<NB * NH, 256, 0, stream>>>(
      y1, l1, u1, wc2, x, lo, up, out, out + NELEM, out + 2 * NELEM);
}
